// Head_29772713296015
// MI455X (gfx1250) — compile-verified, matrix operand order checked by reading, not run on hardware
//
#include <hip/hip_runtime.h>
#include <hip/hip_bf16.h>

typedef __bf16 bf16;
typedef bf16 v16bf __attribute__((ext_vector_type(16)));
typedef bf16 v8bf  __attribute__((ext_vector_type(8)));
typedef float v8f  __attribute__((ext_vector_type(8)));

#define WMMA_BF16(a, b, c) \
  __builtin_amdgcn_wmma_f32_16x16x32_bf16(false, (a), false, (b), (short)0, (c), false, false)

// 16-bit A/B fragment layout (wave32): lane l -> row/col = l%16, half = l/16,
// element i -> k = 16*(i>>3) + 8*half + (i&7). Two contiguous 8-elem (16B) chunks
// at p and p+16 (caller passes p = base + row*ld + k0 + 8*half).
__device__ __forceinline__ v16bf load_frag_bf16(const bf16* p) {
  const v8bf* q = (const v8bf*)p;
  v8bf lo = q[0];
  v8bf hi = q[2];  // +16 elements
  v16bf r;
#pragma unroll
  for (int i = 0; i < 8; ++i) { r[i] = lo[i]; r[i + 8] = hi[i]; }
  return r;
}

__device__ __forceinline__ v16bf load_frag_f32(const float* p) {
  v8f lo = *(const v8f*)p;
  v8f hi = *(const v8f*)(p + 16);
  v16bf r;
#pragma unroll
  for (int i = 0; i < 8; ++i) { r[i] = (bf16)lo[i]; r[i + 8] = (bf16)hi[i]; }
  return r;
}

// ---------------------------------------------------------------------------
// Kernel 1: convert W[C,H] f32 -> Wt[m][H][C] bf16 (transposed). Q scaled by H^-0.5.
// ---------------------------------------------------------------------------
__global__ void hk_prep_weights(const float* __restrict__ Wk,
                                const float* __restrict__ Wq,
                                const float* __restrict__ Wv,
                                bf16* __restrict__ Wt, int C, int H, float qscale) {
  int idx = blockIdx.x * blockDim.x + threadIdx.x;  // over 3*H*C
  int m = idx / (C * H);
  int rem = idx - m * C * H;
  int h = rem / C;
  int c = rem - h * C;
  const float* W = (m == 0) ? Wk : ((m == 1) ? Wq : Wv);
  float val = W[c * H + h];
  if (m == 1) val *= qscale;
  Wt[idx] = (bf16)val;
}

// ---------------------------------------------------------------------------
// Kernel 2: projections. grid.y selects matrix (0=K row-major, 1=Q row-major,
// 2=V transposed [H][T]). One wave per 16-row tile, full H=64 (4 N-tiles),
// K-loop over C in steps of 32.
// ---------------------------------------------------------------------------
__global__ void hk_proj(const float* __restrict__ x, const bf16* __restrict__ Wt,
                        bf16* __restrict__ Kb, bf16* __restrict__ Qb,
                        bf16* __restrict__ Vt, int C, int T) {
  int lane = threadIdx.x & 31;
  int wave = threadIdx.x >> 5;
  int lr = lane & 15, lh = lane >> 4;
  int tile = blockIdx.x * 4 + wave;      // 16-row tile over flat B*T rows
  int m = blockIdx.y;
  int row0 = tile * 16;

  const bf16* W = Wt + (size_t)m * 64 * C;                   // [64][C]
  const float* xrow = x + (size_t)(row0 + lr) * C + 8 * lh;  // A-frag base
  v8f acc[4] = {};

  for (int k0 = 0; k0 < C; k0 += 32) {
    v16bf a = load_frag_f32(xrow + k0);
#pragma unroll
    for (int nt = 0; nt < 4; ++nt) {
      v16bf b = load_frag_bf16(W + (size_t)(nt * 16 + lr) * C + k0 + 8 * lh);
      acc[nt] = WMMA_BF16(a, b, acc[nt]);
    }
  }

  // C-layout: element v -> row (v + 8*lh), column (nt*16 + lr)
  if (m == 2) {  // V transposed: Vt[(b*64 + h)*T + t], t contiguous per lane
    int bb = row0 / T;
    int t0 = (row0 % T) + 8 * lh;
#pragma unroll
    for (int nt = 0; nt < 4; ++nt) {
      int h = nt * 16 + lr;
      v8bf outv;
#pragma unroll
      for (int v = 0; v < 8; ++v) outv[v] = (bf16)acc[nt][v];
      *(v8bf*)(Vt + ((size_t)bb * 64 + h) * T + t0) = outv;
    }
  } else {
    bf16* out = (m == 0) ? Kb : Qb;  // row-major [B*T][64]
#pragma unroll
    for (int nt = 0; nt < 4; ++nt) {
      int h = nt * 16 + lr;
#pragma unroll
      for (int v = 0; v < 8; ++v)
        out[(size_t)(row0 + v + 8 * lh) * 64 + h] = (bf16)acc[nt][v];
    }
  }
}

// ---------------------------------------------------------------------------
// Kernel 3: causal flash attention. Block = 4 waves = 64 query rows; each wave
// owns 16 rows and loops key blocks of 32 up to its own causal bound (no block
// barriers; per-wave LDS tile with explicit s_wait_dscnt).
// ---------------------------------------------------------------------------
__global__ void hk_attn(const bf16* __restrict__ Qb, const bf16* __restrict__ Kb,
                        const bf16* __restrict__ Vt, float* __restrict__ out, int T) {
  __shared__ bf16 Plds[4][16 * 32];
  int lane = threadIdx.x & 31;
  int wave = threadIdx.x >> 5;
  int lr = lane & 15, lh = lane >> 4;
  int tilesPerBatch = T / 64;
  int b = blockIdx.x / tilesPerBatch;
  int qt = blockIdx.x - b * tilesPerBatch;
  int q0 = qt * 64 + wave * 16;  // first query row (within batch) of this wave

  const bf16* Q = Qb + (size_t)b * T * 64;
  const bf16* K = Kb + (size_t)b * T * 64;
  const bf16* V = Vt + (size_t)b * 64 * T;

  v16bf aq0 = load_frag_bf16(Q + (size_t)(q0 + lr) * 64 + 0 + 8 * lh);
  v16bf aq1 = load_frag_bf16(Q + (size_t)(q0 + lr) * 64 + 32 + 8 * lh);

  v8f o[4] = {};
  float m_i[8], l_i[8];
#pragma unroll
  for (int v = 0; v < 8; ++v) { m_i[v] = -__builtin_inff(); l_i[v] = 0.0f; }
  bf16* P = &Plds[wave][0];

  for (int s0 = 0; s0 < q0 + 16; s0 += 32) {
    if (s0 + 32 < q0 + 16)  // prefetch next K block (global_prefetch_b8)
      __builtin_prefetch(K + (size_t)(s0 + 32 + lr) * 64, 0, 0);

    // S chunk 16x32: two 16x16 N-tiles, K-dim = H = 64 (2 WMMAs each)
    v8f sc[2];
#pragma unroll
    for (int nt = 0; nt < 2; ++nt) {
      v8f c = {};
      v16bf bk0 = load_frag_bf16(K + (size_t)(s0 + nt * 16 + lr) * 64 + 0 + 8 * lh);
      c = WMMA_BF16(aq0, bk0, c);
      v16bf bk1 = load_frag_bf16(K + (size_t)(s0 + nt * 16 + lr) * 64 + 32 + 8 * lh);
      sc[nt] = WMMA_BF16(aq1, bk1, c);
    }

    // online softmax: row (v + 8*lh) spread over the 16 lanes of this half
#pragma unroll
    for (int v = 0; v < 8; ++v) {
      int row = q0 + v + 8 * lh;
      float s0v = ((s0 + lr) > row) ? -__builtin_inff() : sc[0][v];
      float s1v = ((s0 + 16 + lr) > row) ? -__builtin_inff() : sc[1][v];
      float cm = fmaxf(s0v, s1v);
#pragma unroll
      for (int off = 1; off < 16; off <<= 1) cm = fmaxf(cm, __shfl_xor(cm, off, 32));
      float mnew = fmaxf(m_i[v], cm);
      float p0 = __expf(s0v - mnew);
      float p1 = __expf(s1v - mnew);
      float scale = __expf(m_i[v] - mnew);
      float rs = p0 + p1;
#pragma unroll
      for (int off = 1; off < 16; off <<= 1) rs += __shfl_xor(rs, off, 32);
      l_i[v] = l_i[v] * scale + rs;
      m_i[v] = mnew;
#pragma unroll
      for (int nt = 0; nt < 4; ++nt) o[nt][v] *= scale;
      // C-layout -> LDS row-major [16 rows][32 keys] for A-layout reload
      P[(v + 8 * lh) * 32 + lr] = (bf16)p0;
      P[(v + 8 * lh) * 32 + 16 + lr] = (bf16)p1;
    }
    asm volatile("s_wait_dscnt 0x0" ::: "memory");  // LDS stores visible to wave

    // O += P(16x32) x V(32x64): A from LDS, B from transposed V (contiguous)
    v16bf ap = load_frag_bf16(P + lr * 32 + 8 * lh);
#pragma unroll
    for (int nt = 0; nt < 4; ++nt) {
      v16bf bv = load_frag_bf16(V + (size_t)(nt * 16 + lr) * T + s0 + 8 * lh);
      o[nt] = WMMA_BF16(ap, bv, o[nt]);
    }
    asm volatile("s_wait_dscnt 0x0" ::: "memory");  // drain P reads before overwrite
  }

#pragma unroll
  for (int nt = 0; nt < 4; ++nt)
#pragma unroll
    for (int v = 0; v < 8; ++v)
      out[((size_t)b * T + q0 + v + 8 * lh) * 64 + nt * 16 + lr] = o[nt][v] / l_i[v];
}

// ---------------------------------------------------------------------------
extern "C" void kernel_launch(void* const* d_in, const int* in_sizes, int n_in,
                              void* d_out, int out_size, void* d_ws, size_t ws_size,
                              hipStream_t stream) {
  const float* x  = (const float*)d_in[0];
  const float* Wk = (const float*)d_in[1];
  const float* Wq = (const float*)d_in[2];
  const float* Wv = (const float*)d_in[3];
  const int C = 1024, H = 64, T = 4096;
  const int BT = in_sizes[0] / C;  // 16384
  const int B = BT / T;

  char* ws = (char*)d_ws;
  size_t off = 0;
  bf16* Wt = (bf16*)(ws + off); off += (size_t)3 * H * C * sizeof(bf16);
  off = (off + 255) & ~(size_t)255;
  bf16* Kb = (bf16*)(ws + off); off += (size_t)BT * H * sizeof(bf16);
  bf16* Qb = (bf16*)(ws + off); off += (size_t)BT * H * sizeof(bf16);
  bf16* Vt = (bf16*)(ws + off); off += (size_t)BT * H * sizeof(bf16);

  hk_prep_weights<<<(3 * C * H) / 256, 256, 0, stream>>>(Wk, Wq, Wv, Wt, C, H, 0.125f);

  dim3 pgrid(BT / 64, 3);  // 4 waves/block, 16 rows/wave
  hk_proj<<<pgrid, 128, 0, stream>>>(x, Wt, Kb, Qb, Vt, C, T);

  hk_attn<<<B * (T / 64), 128, 0, stream>>>(Qb, Kb, Vt, (float*)d_out, T);
}
